// EdgeLearning_73839077752908
// MI455X (gfx1250) — matrix operand order repaired and checked by reading, not compiled, not run on hardware
//
#include <hip/hip_runtime.h>

// ---------------------------------------------------------------------------
// Edge MLP: gather + [E,160]x[160,128] + leaky_relu + [E,128]x[128,64]
// CDNA5 / gfx1250, wave32. Both GEMMs computed transposed:
//   h.T  = W1 x mask.T   (A = W1 f16 in LDS, B = gathered edge features)
//   out.T = W2 x h.T     (B = GEMM1 result, fixed up with a half-wave swap)
// so the inter-layer transpose is a register shuffle, not an LDS round-trip.
// A per-iteration memory clobber stops LICM from hoisting the (loop-invariant)
// weight-fragment LDS loads into ~450 live VGPRs and spilling to scratch.
// ---------------------------------------------------------------------------

typedef _Float16 v8h  __attribute__((ext_vector_type(8)));
typedef _Float16 v16h __attribute__((ext_vector_type(16)));
typedef float    v8f  __attribute__((ext_vector_type(8)));
typedef int      v4i  __attribute__((ext_vector_type(4)));

#define DN    64
#define DE    32
#define DIN   160   // 2*DN + DE
#define DH    128
#define DOUT  64

// Row paddings (f16 elems) so the LDS row stride in dwords avoids 64-bank
// conflicts for 16-consecutive-row A-fragment reads: 168*2B=84 dw, 136*2B=68 dw.
#define W1P   168
#define W2P   136

static __device__ inline v16h cat16(v8h lo, v8h hi) {
  return __builtin_shufflevector(lo, hi, 0, 1, 2, 3, 4, 5, 6, 7,
                                         8, 9, 10, 11, 12, 13, 14, 15);
}

extern "C" __global__ __launch_bounds__(256)
void edge_mlp_wmma(const float* __restrict__ x,
                   const int*   __restrict__ eidx,   // [2, E]
                   const float* __restrict__ eattr,  // [E, 32]
                   const float* __restrict__ W1,     // [128, 160] row-major
                   const float* __restrict__ b1,     // [128]
                   const float* __restrict__ W2,     // [64, 128] row-major
                   const float* __restrict__ b2,     // [64]
                   float*       __restrict__ out,    // [E, 64]
                   int E)
{
  __shared__ __align__(16) _Float16 w1h[DH * W1P];    // 43008 B
  __shared__ __align__(16) _Float16 w2h[DOUT * W2P];  // 17408 B
  __shared__ __align__(32) float    b1s[DH];
  __shared__ __align__(32) float    b2s[DOUT];        // total ~61 KB < 64 KB

  const int tid = threadIdx.x;

  // ---- Stage weights f32 -> f16 into LDS (A-fragment friendly, padded rows)
  for (int i = tid; i < DH * DIN; i += 256) {
    int r = i / DIN, c = i - r * DIN;
    w1h[r * W1P + c] = (_Float16)W1[i];
  }
  for (int i = tid; i < DOUT * DH; i += 256) {
    int r = i / DH, c = i - r * DH;
    w2h[r * W2P + c] = (_Float16)W2[i];
  }
  if (tid < DH)   b1s[tid] = b1[tid];
  if (tid < DOUT) b2s[tid] = b2[tid];
  __syncthreads();

  const int lane  = tid & 31;
  const int warp  = tid >> 5;
  const int nlo   = lane & 15;             // A: local row n | B: edge column m
  const int aoff  = (lane < 16) ? 0 : 16;  // A-fragment K chunk offset
  const int khalf = (lane < 16) ? 0 : 16;  // B-fragment K offset
  const int mo    = (lane < 16) ? 0 : 8;   // C/D row (n) offset

  const int tiles  = (E + 15) >> 4;
  const int slot   = blockIdx.x * 8 + warp;
  const int nslots = gridDim.x * 8;

  for (int tile = slot; tile < tiles; tile += nslots) {
    // Opaque barrier: keeps the weight-fragment LDS loads inside the loop
    // (without it LICM hoists 56 v16h fragments -> VGPR spill to scratch).
    asm volatile("" ::: "memory");

    const int e0 = tile << 4;
    const int e  = e0 + nlo;               // this lane's edge row
    const int ec = (e < E) ? e : (E - 1);
    const int si = eidx[ec];
    const int sj = eidx[E + ec];
    const float* pI = x + (long)si * DN;
    const float* pJ = x + (long)sj * DN;
    const float* pE = eattr + (long)ec * DE;

    // ---------------- GEMM1: h.T[n][m] = W1 x mask.T ----------------
    v8f acc1[8] = {};
#pragma unroll
    for (int t = 0; t < 5; ++t) {
      // B fragment: 16 contiguous source floats at k = 32t + khalf
      // (chunk starts are multiples of 16 -> never straddle the 64/128 splits)
      const int k = t * 32 + khalf;
      const float* sp = (k < 64) ? (pI + k)
                      : ((k < 128) ? (pJ + (k - 64)) : (pE + (k - 128)));
      const float4* q = (const float4*)sp;   // 64B-aligned contiguous
      float4 u0 = q[0], u1 = q[1], u2 = q[2], u3 = q[3];
      v16h bf;
      bf[0]  = (_Float16)u0.x; bf[1]  = (_Float16)u0.y;
      bf[2]  = (_Float16)u0.z; bf[3]  = (_Float16)u0.w;
      bf[4]  = (_Float16)u1.x; bf[5]  = (_Float16)u1.y;
      bf[6]  = (_Float16)u1.z; bf[7]  = (_Float16)u1.w;
      bf[8]  = (_Float16)u2.x; bf[9]  = (_Float16)u2.y;
      bf[10] = (_Float16)u2.z; bf[11] = (_Float16)u2.w;
      bf[12] = (_Float16)u3.x; bf[13] = (_Float16)u3.y;
      bf[14] = (_Float16)u3.z; bf[15] = (_Float16)u3.w;
#pragma unroll
      for (int nb = 0; nb < 8; ++nb) {
        const _Float16* wa = w1h + (nb * 16 + nlo) * W1P + t * 32 + aoff;
        v16h af = cat16(*(const v8h*)wa, *(const v8h*)(wa + 8));
        acc1[nb] = __builtin_amdgcn_wmma_f32_16x16x32_f16(
            false, af, false, bf, (short)0, acc1[nb], false, false);
      }
    }

    // ---- bias + leaky_relu + f16 pack + half-wave swap (register transpose)
    // Lane holds h[m][nb*16+mo .. +7]; partner lane (xor 16) holds the other
    // 8 rows of the same column m -> one SWAPX16 shuffle per dword.
    v8h own[8], part[8];
#pragma unroll
    for (int nb = 0; nb < 8; ++nb) {
      v8f bias = *(const v8f*)(b1s + nb * 16 + mo);  // uniform per half-wave
      v8h o;
#pragma unroll
      for (int v = 0; v < 8; ++v) {
        float y = acc1[nb][v] + bias[v];
        y = (y >= 0.0f) ? y : 0.01f * y;
        o[v] = (_Float16)y;
      }
      own[nb] = o;
      v4i oi = __builtin_bit_cast(v4i, o);
      v4i pi;
#pragma unroll
      for (int d = 0; d < 4; ++d) pi[d] = __shfl_xor(oi[d], 16, 32);
      part[nb] = __builtin_bit_cast(v8h, pi);
    }

    // ---------------- GEMM2: out.T[n][m] = W2 x h.T ----------------
    v8f acc2[4] = {};
#pragma unroll
    for (int t = 0; t < 4; ++t) {
      // B fragment for K = 32t + khalf .. +15, assembled from registers:
      v8h blo = (lane < 16) ? own[2 * t]  : part[2 * t + 1];
      v8h bhi = (lane < 16) ? part[2 * t] : own[2 * t + 1];
      v16h bf = cat16(blo, bhi);
#pragma unroll
      for (int nb = 0; nb < 4; ++nb) {
        const _Float16* wa = w2h + (nb * 16 + nlo) * W2P + t * 32 + aoff;
        v16h af = cat16(*(const v8h*)wa, *(const v8h*)(wa + 8));
        acc2[nb] = __builtin_amdgcn_wmma_f32_16x16x32_f16(
            false, af, false, bf, (short)0, acc2[nb], false, false);
      }
    }

    // ---- bias + store: lane owns half of its edge row, 8 contiguous cols per
    // nb tile -> wide 32B stores.
    if (e < E) {
      float* po = out + (long)e * DOUT;
#pragma unroll
      for (int nb = 0; nb < 4; ++nb) {
        v8f bias = *(const v8f*)(b2s + nb * 16 + mo);
        v8f r;
#pragma unroll
        for (int v = 0; v < 8; ++v) r[v] = acc2[nb][v] + bias[v];
        *(v8f*)(po + nb * 16 + mo) = r;
      }
    }
  }
}

extern "C" void kernel_launch(void* const* d_in, const int* in_sizes, int n_in,
                              void* d_out, int out_size, void* d_ws, size_t ws_size,
                              hipStream_t stream) {
  const float* x   = (const float*)d_in[0];
  const int*   ei  = (const int*)d_in[1];
  const float* ea  = (const float*)d_in[2];
  const float* W1  = (const float*)d_in[3];
  const float* b1  = (const float*)d_in[4];
  const float* W2  = (const float*)d_in[5];
  const float* b2  = (const float*)d_in[6];
  float* out = (float*)d_out;
  const int E = in_sizes[1] / 2;  // edge_index is [2, E]

  const int blocks = 625;  // 625 * 8 wave-slots -> 10 tiles per wave for E=800k
  edge_mlp_wmma<<<dim3(blocks), dim3(256), 0, stream>>>(
      x, ei, ea, W1, b1, W2, b2, out, E);
}
